// Model_1597727834173
// MI455X (gfx1250) — compile-verified, matrix operand order checked by reading, not run on hardware
//
#include <hip/hip_runtime.h>

#define B_   64
#define T_   2048
#define E_   256
#define KW_  3
#define NT_  32
#define TP_  (T_ + 2)   // halo rows at t=0 and t=T+1

typedef __attribute__((ext_vector_type(16))) __bf16 v16bf;
typedef __attribute__((ext_vector_type(8)))  __bf16 v8bf;
typedef __attribute__((ext_vector_type(8)))  float  v8f;
typedef __attribute__((ext_vector_type(4)))  float  v4f;

// ---------------------------------------------------------------------------
// Zero the t-halo rows of both ping-pong activation buffers.
// threads = 2 * B * 2 * 256 = 65536
__global__ void k_zero_halo(__bf16* __restrict__ Xa, __bf16* __restrict__ Xb) {
    int tid = blockIdx.x * blockDim.x + threadIdx.x;
    int buf = tid >> 15;            // 0 = Xa, 1 = Xb
    int rem = tid & 32767;
    int b   = rem >> 9;             // 64
    int rc  = rem & 511;
    int r   = (rc >> 8) ? (T_ + 1) : 0;
    int c   = rc & 255;
    __bf16* X = buf ? Xb : Xa;
    X[((size_t)b * TP_ + r) * E_ + c] = (__bf16)0.0f;
}

// ---------------------------------------------------------------------------
// Embedding gather: fp32 table -> bf16 [B][T+2][C] (rows 1..T).
// threads = B * T * (E/8) = 4194304
__global__ void k_gather(const int* __restrict__ tok,
                         const float* __restrict__ emb,
                         __bf16* __restrict__ Xa) {
    int tid = blockIdx.x * blockDim.x + threadIdx.x;
    int c8  = tid & 31;
    int t   = (tid >> 5) & (T_ - 1);
    int b   = tid >> 16;
    int tk  = tok[b * T_ + t];
    const float* src = emb + (size_t)tk * E_ + c8 * 8;
    v4f f0 = *(const v4f*)(src);
    v4f f1 = *(const v4f*)(src + 4);
    v8bf o;
    o[0] = (__bf16)f0[0]; o[1] = (__bf16)f0[1];
    o[2] = (__bf16)f0[2]; o[3] = (__bf16)f0[3];
    o[4] = (__bf16)f1[0]; o[5] = (__bf16)f1[1];
    o[6] = (__bf16)f1[2]; o[7] = (__bf16)f1[3];
    *(v8bf*)(Xa + ((size_t)b * TP_ + t + 1) * E_ + c8 * 8) = o;
}

// ---------------------------------------------------------------------------
// Prepack conv weights as WMMA A-fragments (bf16) and fc weights as
// B-fragments. Fragment = 32 lanes x 16 elems, lane-major.
// A/B 16-bit 16x32 lane layout: lane<16 -> row = lane, K in {0..7,16..23};
// lane>=16 -> row = lane-16, K in {8..15,24..31}; element e -> K = kh + (e<8?e:e+8)
// conv pack index: frag f = (ct*3 + s)*8 + cc ; fc pack: f = nt*8 + cc
#define CONVPACK_N (16 * 3 * 8 * 512)   // 196608 elems
#define FCPACK_N   (16 * 512)           // 8192 elems
__global__ void k_prepack(const float* __restrict__ w1,
                          const float* __restrict__ w2,
                          const float* __restrict__ fw,
                          __bf16* __restrict__ p1,
                          __bf16* __restrict__ p2,
                          __bf16* __restrict__ pf) {
    int tid = blockIdx.x * blockDim.x + threadIdx.x;
    if (tid < 2 * CONVPACK_N) {
        const float* w = (tid < CONVPACK_N) ? w1 : w2;
        __bf16* p      = (tid < CONVPACK_N) ? p1 : p2;
        int idx  = (tid < CONVPACK_N) ? tid : tid - CONVPACK_N;
        int e    = idx & 15;
        int lane = (idx >> 4) & 31;
        int f    = idx >> 9;
        int cc   = f & 7;
        int s    = (f >> 3) % 3;
        int ct   = f / 24;
        int row  = ct * 16 + (lane & 15);                              // c_out
        int kk   = cc * 32 + ((lane >> 4) << 3) + (e < 8 ? e : e + 8); // c_in
        p[idx] = (__bf16)w[(size_t)row * (E_ * KW_) + (size_t)kk * KW_ + s];
    } else {
        int idx = tid - 2 * CONVPACK_N;
        if (idx < FCPACK_N) {
            int e    = idx & 15;
            int lane = (idx >> 4) & 31;
            int f    = idx >> 9;             // 0..15
            int cc   = f & 7;
            int nt   = f >> 3;
            int tag  = nt * 16 + (lane & 15);
            int kk   = cc * 32 + ((lane >> 4) << 3) + (e < 8 ? e : e + 8);
            pf[idx] = (__bf16)fw[(size_t)tag * E_ + kk];
        }
    }
}

// ---------------------------------------------------------------------------
// Conv1d(256->256, k=3, pad=1) as 3 shifted GEMMs via WMMA bf16.
// Register macro-tile: one wave computes 64(c_out) x 64(t):
//   16 f32 accumulators; per K-step 4 A-frags + 4 B-frags -> 16 WMMAs
//   (32 FLOP/byte of fragment traffic, 4x the naive 16x16 tiling).
// waves = B * (T/64) * (C/64) = 8192 ; block = 256 (8 waves) -> 1024 blocks
__global__ void __launch_bounds__(256)
k_conv(const __bf16* __restrict__ Xin,   // [B][T+2][C]
       const __bf16* __restrict__ Wp,    // packed A-frags
       const float*  __restrict__ bias,
       __bf16* __restrict__ Xout) {
    int gtid = blockIdx.x * 256 + threadIdx.x;
    int lane = threadIdx.x & 31;
    int wave = gtid >> 5;
    int ctg  = wave & 3;            // 4 groups of 64 c_out
    int ttg  = (wave >> 2) & 31;    // 32 groups of 64 t
    int b    = wave >> 7;           // 64
    int t0   = ttg << 6;
    int coB  = ctg << 6;
    int l15  = lane & 15;
    int kh   = (lane >> 4) << 3;    // 0 or 8

    v8f acc[4][4];
#pragma unroll
    for (int m = 0; m < 4; ++m)
#pragma unroll
        for (int j = 0; j < 4; ++j) acc[m][j] = (v8f){};

    const __bf16* xb = Xin + (size_t)b * TP_ * E_;
#pragma unroll
    for (int s = 0; s < 3; ++s) {
#pragma unroll
        for (int cc = 0; cc < 8; ++cc) {
            v16bf A[4], Bf[4];
#pragma unroll
            for (int m = 0; m < 4; ++m) {
                int ct = (ctg << 2) + m;
                A[m] = *(const v16bf*)(Wp + (size_t)((ct * 3 + s) * 8 + cc) * 512
                                          + lane * 16);
            }
#pragma unroll
            for (int j = 0; j < 4; ++j) {
                const __bf16* xr = xb + (size_t)(t0 + (j << 4) + l15 + s) * E_
                                      + cc * 32 + kh;
                v8bf lo = *(const v8bf*)(xr);
                v8bf hi = *(const v8bf*)(xr + 16);
                Bf[j] = __builtin_shufflevector(lo, hi, 0, 1, 2, 3, 4, 5, 6, 7,
                                                8, 9, 10, 11, 12, 13, 14, 15);
            }
#pragma unroll
            for (int m = 0; m < 4; ++m)
#pragma unroll
                for (int j = 0; j < 4; ++j)
                    acc[m][j] = __builtin_amdgcn_wmma_f32_16x16x32_bf16(
                        false, A[m], false, Bf[j], (short)0, acc[m][j],
                        false, false);
        }
    }
    // D layout: VGPR v, lane n<16 -> (M=v, N=n); lane>=16 -> (M=v+8, N=n-16).
    // A lane's 8 acc entries are contiguous in c_out -> one 16B bf16 store.
#pragma unroll
    for (int m = 0; m < 4; ++m) {
        int co0 = coB + (m << 4) + kh;
        v4f bz0 = *(const v4f*)(bias + co0);
        v4f bz1 = *(const v4f*)(bias + co0 + 4);
#pragma unroll
        for (int j = 0; j < 4; ++j) {
            __bf16* orow = Xout + ((size_t)b * TP_ + (t0 + (j << 4) + l15 + 1)) * E_
                                + co0;
            v8bf o;
            o[0] = (__bf16)(acc[m][j][0] + bz0[0]);
            o[1] = (__bf16)(acc[m][j][1] + bz0[1]);
            o[2] = (__bf16)(acc[m][j][2] + bz0[2]);
            o[3] = (__bf16)(acc[m][j][3] + bz0[3]);
            o[4] = (__bf16)(acc[m][j][4] + bz1[0]);
            o[5] = (__bf16)(acc[m][j][5] + bz1[1]);
            o[6] = (__bf16)(acc[m][j][6] + bz1[2]);
            o[7] = (__bf16)(acc[m][j][7] + bz1[3]);
            *(v8bf*)orow = o;
        }
    }
}

// ---------------------------------------------------------------------------
// FC: logits[b][t][tag] = h . fc_w[tag] + fc_b  via WMMA (M=t, N=tag, K=256).
// Macro-tile 64(t) x 32(tags): 8 accumulators, 64 WMMAs per wave.
// waves = B*(T/64) = 2048 -> 256 blocks
__global__ void __launch_bounds__(256)
k_fc(const __bf16* __restrict__ H,      // [B][T+2][C]
     const __bf16* __restrict__ Fp,     // packed B-frags
     const float*  __restrict__ fb,
     float* __restrict__ logits) {      // [B][T][32]
    int gtid = blockIdx.x * 256 + threadIdx.x;
    int lane = threadIdx.x & 31;
    int wave = gtid >> 5;
    int ttg  = wave & 31;
    int b    = wave >> 5;
    int t0   = ttg << 6;
    int l15  = lane & 15;
    int kh   = (lane >> 4) << 3;

    v8f acc[4][2];
#pragma unroll
    for (int m = 0; m < 4; ++m) {
        acc[m][0] = (v8f){};
        acc[m][1] = (v8f){};
    }
#pragma unroll
    for (int cc = 0; cc < 8; ++cc) {
        v16bf b0 = *(const v16bf*)(Fp + (size_t)cc * 512 + lane * 16);
        v16bf b1 = *(const v16bf*)(Fp + (size_t)(8 + cc) * 512 + lane * 16);
#pragma unroll
        for (int m = 0; m < 4; ++m) {
            const __bf16* hr = H + ((size_t)b * TP_ + (t0 + (m << 4) + l15 + 1)) * E_
                                 + cc * 32 + kh;
            v8bf  lo = *(const v8bf*)(hr);
            v8bf  hi = *(const v8bf*)(hr + 16);
            v16bf aa = __builtin_shufflevector(lo, hi, 0, 1, 2, 3, 4, 5, 6, 7,
                                               8, 9, 10, 11, 12, 13, 14, 15);
            acc[m][0] = __builtin_amdgcn_wmma_f32_16x16x32_bf16(
                false, aa, false, b0, (short)0, acc[m][0], false, false);
            acc[m][1] = __builtin_amdgcn_wmma_f32_16x16x32_bf16(
                false, aa, false, b1, (short)0, acc[m][1], false, false);
        }
    }
    float* lr = logits + ((size_t)b * T_ + t0) * NT_;
#pragma unroll
    for (int m = 0; m < 4; ++m)
#pragma unroll
        for (int v = 0; v < 8; ++v) {
            int tl = (m << 4) + v + kh;
            lr[(size_t)tl * NT_ + l15]      = acc[m][0][v] + fb[l15];
            lr[(size_t)tl * NT_ + 16 + l15] = acc[m][1][v] + fb[16 + l15];
        }
}

// ---------------------------------------------------------------------------
// CRF log-partition, exp-domain matvec formulation (exact):
//   logsumexp_i(a_i + t_ij) = m + log( sum_i exp(a_i - m) * exp(t_ij) )
// One wave (32 lanes = 32 tags) per batch row.
__device__ __forceinline__ float wave_max(float v) {
    v = fmaxf(v, __shfl_xor(v, 1, 32));
    v = fmaxf(v, __shfl_xor(v, 2, 32));
    v = fmaxf(v, __shfl_xor(v, 4, 32));
    v = fmaxf(v, __shfl_xor(v, 8, 32));
    v = fmaxf(v, __shfl_xor(v, 16, 32));
    return v;
}
__device__ __forceinline__ float wave_sum(float v) {
    v += __shfl_xor(v, 1, 32);
    v += __shfl_xor(v, 2, 32);
    v += __shfl_xor(v, 4, 32);
    v += __shfl_xor(v, 8, 32);
    v += __shfl_xor(v, 16, 32);
    return v;
}

__global__ void k_crf(const float* __restrict__ logits,   // [B][T][32]
                      const float* __restrict__ trans,    // [32][32]
                      const float* __restrict__ start_t,
                      const float* __restrict__ end_t,
                      float* __restrict__ out) {          // [B]
    int b = blockIdx.x;
    int j = threadIdx.x;           // tag
    float et[32];
#pragma unroll
    for (int i = 0; i < 32; ++i) et[i] = __expf(trans[i * 32 + j]);

    const float* em = logits + (size_t)b * T_ * NT_;
    float alpha = start_t[j] + em[j];
    for (int t = 1; t < T_; ++t) {
        em += NT_;
        float m = wave_max(alpha);
        float p = __expf(alpha - m);
        float q = 0.0f;
#pragma unroll
        for (int i = 0; i < 32; ++i)
            q = fmaf(__shfl(p, i, 32), et[i], q);
        alpha = m + __logf(q) + em[j];
    }
    float v  = alpha + end_t[j];
    float m2 = wave_max(v);
    float s  = wave_sum(__expf(v - m2));
    if (j == 0) out[b] = m2 + __logf(s);
}

// ---------------------------------------------------------------------------
extern "C" void kernel_launch(void* const* d_in, const int* in_sizes, int n_in,
                              void* d_out, int out_size, void* d_ws, size_t ws_size,
                              hipStream_t stream) {
    const int*   tok     = (const int*)  d_in[0];
    const float* emb     = (const float*)d_in[1];
    const float* conv1_w = (const float*)d_in[2];
    const float* conv1_b = (const float*)d_in[3];
    const float* conv2_w = (const float*)d_in[4];
    const float* conv2_b = (const float*)d_in[5];
    const float* fc_w    = (const float*)d_in[6];
    const float* fc_b    = (const float*)d_in[7];
    const float* trans   = (const float*)d_in[8];
    const float* start_t = (const float*)d_in[9];
    const float* end_t   = (const float*)d_in[10];
    float* out = (float*)d_out;

    // workspace layout (all offsets multiples of 256B)
    const size_t szX   = (size_t)B_ * TP_ * E_ * sizeof(__bf16);   // 67,174,400
    const size_t szLog = (size_t)B_ * T_ * NT_ * sizeof(float);    // 16,777,216
    const size_t szWp  = (size_t)CONVPACK_N * sizeof(__bf16);      //    393,216
    char* ws = (char*)d_ws;
    __bf16* Xa  = (__bf16*)(ws);
    __bf16* Xb  = (__bf16*)(ws + szX);
    float*  lg  = (float*) (ws + 2 * szX);
    __bf16* W1p = (__bf16*)(ws + 2 * szX + szLog);
    __bf16* W2p = (__bf16*)(ws + 2 * szX + szLog + szWp);
    __bf16* FCp = (__bf16*)(ws + 2 * szX + szLog + 2 * szWp);

    // 1. zero halo rows of both activation buffers
    k_zero_halo<<<256, 256, 0, stream>>>(Xa, Xb);
    // 2. prepack weights into WMMA fragment order
    k_prepack<<<(2 * CONVPACK_N + FCPACK_N) / 256, 256, 0, stream>>>(
        conv1_w, conv2_w, fc_w, W1p, W2p, FCp);
    // 3. embedding gather -> Xa
    k_gather<<<(B_ * T_ * (E_ / 8)) / 256, 256, 0, stream>>>(tok, emb, Xa);
    // 4. conv1: Xa -> Xb   (8192 waves, 64x64 macro-tiles)
    k_conv<<<(B_ * (T_ / 64) * (E_ / 64) * 32) / 256, 256, 0, stream>>>(
        Xa, W1p, conv1_b, Xb);
    // 5. conv2: Xb -> Xa
    k_conv<<<(B_ * (T_ / 64) * (E_ / 64) * 32) / 256, 256, 0, stream>>>(
        Xb, W2p, conv2_b, Xa);
    // 6. FC -> logits (2048 waves)
    k_fc<<<(B_ * (T_ / 64) * 32) / 256, 256, 0, stream>>>(Xa, FCp, fc_b, lg);
    // 7. CRF scan -> out[B]
    k_crf<<<B_, 32, 0, stream>>>(lg, trans, start_t, end_t, out);
    (void)in_sizes; (void)n_in; (void)out_size; (void)ws_size;
}
